// Self_Attention_Alibi_90091234001517
// MI455X (gfx1250) — matrix operand order repaired and checked by reading, not compiled, not run on hardware
//
#include <hip/hip_runtime.h>

#define BQ  4
#define SQ  2048
#define HQ  1024
#define NHQ 16
#define DQ  64

typedef __attribute__((ext_vector_type(16))) __bf16 v16bf;
typedef __attribute__((ext_vector_type(8)))  __bf16 v8bf;
typedef __attribute__((ext_vector_type(4)))  __bf16 v4bf;
typedef __attribute__((ext_vector_type(8)))  float  v8f;

static __device__ inline v8f wmma_bf16(v16bf a, v16bf b, v8f c) {
  // D = A(16x32 bf16) * B(32x16 bf16) + C(16x16 f32)
  return __builtin_amdgcn_wmma_f32_16x16x32_bf16(
      /*neg_a=*/false, a, /*neg_b=*/false, b,
      /*c_mod=*/(short)0, c, /*reuse_a=*/false, /*reuse_b=*/false);
}

// A-matrix (16x32, MxK) fragment from row-major [row][K], leading dim ld.
// lane<16: row=lane, K={0..7}U{16..23}; lane>=16: row=lane-16, K={8..15}U{24..31}
static __device__ inline v16bf load_a_frag(const __bf16* base, int row0, int ld,
                                           int k0, int lane) {
  int r  = lane & 15;
  int kb = (lane >> 4) << 3;  // 0 or 8
  const __bf16* p = base + (size_t)(row0 + r) * ld + (k0 + kb);
  v8bf lo = *(const v8bf*)(p);
  v8bf hi = *(const v8bf*)(p + 16);
  return __builtin_shufflevector(lo, hi, 0,1,2,3,4,5,6,7,8,9,10,11,12,13,14,15);
}

// B-matrix (32x16, KxN) fragment from [N][K] row-major storage, leading dim ld.
// lane<16: col=lane, K={0..7}U{16..23}; lane>=16: col=lane-16, K={8..15}U{24..31}
static __device__ inline v16bf load_b_frag(const __bf16* base, int n0, int ld,
                                           int k0, int lane) {
  int nn = lane & 15;
  int kb = (lane >> 4) << 3;  // 0 or 8
  const __bf16* p = base + (size_t)(n0 + nn) * ld + (k0 + kb);
  v8bf lo = *(const v8bf*)(p);
  v8bf hi = *(const v8bf*)(p + 16);
  return __builtin_shufflevector(lo, hi, 0,1,2,3,4,5,6,7,8,9,10,11,12,13,14,15);
}

// ---------------- fp32 -> bf16 convert ----------------
__global__ __launch_bounds__(256) void cvt_f32_bf16_kernel(
    const float* __restrict__ in, __bf16* __restrict__ out, int n4) {
  int i = blockIdx.x * blockDim.x + threadIdx.x;
  if (i < n4) {
    float4 f = ((const float4*)in)[i];
    v4bf o;
    o[0] = (__bf16)f.x; o[1] = (__bf16)f.y;
    o[2] = (__bf16)f.z; o[3] = (__bf16)f.w;
    ((v4bf*)out)[i] = o;
  }
}

// ---------------- QKV projection (fused Q,K,V per head) ----------------
// grid.x: (B*S)/128 row tiles; grid.y: head. Each wave: 16 rows x 64 cols x
// 3 mats -> one A fragment feeds 12 WMMAs.  B fragments are batched into
// independent temporaries so loads clause and WMMAs overlap outstanding loads.
__global__ __launch_bounds__(256) void qkv_proj_kernel(
    const __bf16* __restrict__ xb,
    const __bf16* __restrict__ wqb, const __bf16* __restrict__ wkb,
    const __bf16* __restrict__ wvb,
    __bf16* __restrict__ Qb, __bf16* __restrict__ Kb, __bf16* __restrict__ Vt) {
  int lane = threadIdx.x & 31;
  int wave = threadIdx.x >> 5;
  int head = blockIdx.y;
  int grow = blockIdx.x * 128 + wave * 16;   // row within [0, B*S)
  int b = grow / SQ;
  int s = grow % SQ;

  const __bf16* W0 = wqb + (size_t)head * DQ * HQ;
  const __bf16* W1 = wkb + (size_t)head * DQ * HQ;
  const __bf16* W2 = wvb + (size_t)head * DQ * HQ;

  v8f acc[12] = {};  // [mat*4 + j]
  for (int kk = 0; kk < HQ; kk += 32) {
    v16bf a = load_a_frag(xb, grow, HQ, kk, lane);
#pragma unroll
    for (int jj = 0; jj < 4; jj += 2) {
      // batch 6 independent B fragments, then 6 back-to-back WMMAs
      v16bf b00 = load_b_frag(W0, (jj + 0) * 16, HQ, kk, lane);
      v16bf b10 = load_b_frag(W1, (jj + 0) * 16, HQ, kk, lane);
      v16bf b20 = load_b_frag(W2, (jj + 0) * 16, HQ, kk, lane);
      v16bf b01 = load_b_frag(W0, (jj + 1) * 16, HQ, kk, lane);
      v16bf b11 = load_b_frag(W1, (jj + 1) * 16, HQ, kk, lane);
      v16bf b21 = load_b_frag(W2, (jj + 1) * 16, HQ, kk, lane);
      acc[0 * 4 + jj]     = wmma_bf16(a, b00, acc[0 * 4 + jj]);
      acc[1 * 4 + jj]     = wmma_bf16(a, b10, acc[1 * 4 + jj]);
      acc[2 * 4 + jj]     = wmma_bf16(a, b20, acc[2 * 4 + jj]);
      acc[0 * 4 + jj + 1] = wmma_bf16(a, b01, acc[0 * 4 + jj + 1]);
      acc[1 * 4 + jj + 1] = wmma_bf16(a, b11, acc[1 * 4 + jj + 1]);
      acc[2 * 4 + jj + 1] = wmma_bf16(a, b21, acc[2 * 4 + jj + 1]);
    }
  }

  int n   = lane & 15;
  int mhi = (lane >> 4) << 3;  // C layout: VGPR i -> row i (+8 for hi lanes)
  {  // Q: [b][head][s][d]
    __bf16* outp = Qb + (((size_t)b * NHQ + head) * SQ + s) * DQ;
#pragma unroll
    for (int j = 0; j < 4; ++j)
#pragma unroll
      for (int i = 0; i < 8; ++i)
        outp[(size_t)(i + mhi) * DQ + (j * 16 + n)] = (__bf16)acc[0 * 4 + j][i];
  }
  {  // K: [b][head][s][d]
    __bf16* outp = Kb + (((size_t)b * NHQ + head) * SQ + s) * DQ;
#pragma unroll
    for (int j = 0; j < 4; ++j)
#pragma unroll
      for (int i = 0; i < 8; ++i)
        outp[(size_t)(i + mhi) * DQ + (j * 16 + n)] = (__bf16)acc[1 * 4 + j][i];
  }
  {  // V transposed: [b][head][d][s]
    __bf16* outp = Vt + ((size_t)b * NHQ + head) * DQ * SQ;
#pragma unroll
    for (int j = 0; j < 4; ++j)
#pragma unroll
      for (int i = 0; i < 8; ++i)
        outp[(size_t)(j * 16 + n) * SQ + (s + i + mhi)] = (__bf16)acc[2 * 4 + j][i];
  }
}

// ---------------- flash attention with ALiBi ----------------
// grid = (S/128, NH, B). All 8 waves share (b,head): K/V tiles are staged once
// per workgroup into LDS via async copies (ASYNCcnt), double-buffered so the
// next tile's DMA overlaps the current tile's WMMAs + softmax.
__global__ __launch_bounds__(256) void attn_alibi_kernel(
    const __bf16* __restrict__ Qb, const __bf16* __restrict__ Kb,
    const __bf16* __restrict__ Vt, __bf16* __restrict__ Ob) {
  __shared__ __bf16 Ks[2][32][DQ];   // keys tile  [s_local][d]   2 x 4KB
  __shared__ __bf16 Vs[2][DQ][32];   // values tile [d][s_local]  2 x 4KB
  __shared__ __bf16 Pt[8 * 16 * 32]; // per-wave P staging        8KB

  int tid  = threadIdx.x;
  int lane = tid & 31;
  int wave = tid >> 5;
  int b    = blockIdx.z;
  int head = blockIdx.y;
  int q0   = blockIdx.x * 128 + wave * 16;

  const __bf16* Qh = Qb + ((size_t)b * NHQ + head) * SQ * DQ;
  const __bf16* Kh = Kb + ((size_t)b * NHQ + head) * SQ * DQ;
  const __bf16* Vh = Vt + ((size_t)b * NHQ + head) * DQ * SQ;
  __bf16* pt = Pt + wave * (16 * 32);

  const float slope = -(float)(head + 1);
  const float inv_sqrt_d = 0.125f;  // 1/sqrt(64)

  // Per-thread async staging: thread `tid` moves one 16B chunk of the K tile
  // and one 16B chunk of the V tile (256 threads x 16B = 4KB each).
  // K tile is contiguous at Kh + kt*64.  V chunk: d=tid/4, s0=(tid%4)*8.
  int vd = tid >> 2, vs0 = (tid & 3) << 3;
#define STAGE_TILES(buf, kt)                                                   \
  do {                                                                         \
    const char* kg = (const char*)(Kh + (size_t)(kt) * DQ) + tid * 16;         \
    unsigned kl = (unsigned)(size_t)(&Ks[(buf)][0][0]) + tid * 16;             \
    asm volatile("global_load_async_to_lds_b128 %0, %1, off" ::"v"(kl),        \
                 "v"(kg)                                                       \
                 : "memory");                                                  \
    const char* vg = (const char*)(Vh + (size_t)vd * SQ + (kt) + vs0);         \
    unsigned vl = (unsigned)(size_t)(&Vs[(buf)][0][0]) + tid * 16;             \
    asm volatile("global_load_async_to_lds_b128 %0, %1, off" ::"v"(vl),        \
                 "v"(vg)                                                       \
                 : "memory");                                                  \
  } while (0)

  // Q fragments resident across the whole key stream (d chunks 0..31 / 32..63)
  v16bf qa0 = load_a_frag(Qh, q0, DQ, 0, lane);
  v16bf qa1 = load_a_frag(Qh, q0, DQ, 32, lane);

  v8f acc[4] = {};
  float mrow[8], lrow[8];
#pragma unroll
  for (int i = 0; i < 8; ++i) { mrow[i] = -3.0e38f; lrow[i] = 0.0f; }

  int n   = lane & 15;
  int mhi = (lane >> 4) << 3;

  STAGE_TILES(0, 0);  // prime the pipeline

  for (int kt = 0; kt < SQ; kt += 32) {
    int buf = (kt >> 5) & 1;
    bool has_next = (kt + 32) < SQ;
    if (has_next) {
      STAGE_TILES(buf ^ 1, kt + 32);  // overlap next tile DMA with compute
      asm volatile("s_wait_asynccnt 2" ::: "memory");  // current tile done
    } else {
      asm volatile("s_wait_asynccnt 0" ::: "memory");
    }
    __syncthreads();  // all waves' staging visible

    const __bf16* ks = &Ks[buf][0][0];
    const __bf16* vs = &Vs[buf][0][0];

    // ---- scores: batch 4 K fragments from LDS, then 4 WMMAs ----
    v8f s0 = {}, s1 = {};
    {
      v16bf kb00 = load_b_frag(ks, 0, DQ, 0, lane);
      v16bf kb01 = load_b_frag(ks, 0, DQ, 32, lane);
      v16bf kb10 = load_b_frag(ks, 16, DQ, 0, lane);
      v16bf kb11 = load_b_frag(ks, 16, DQ, 32, lane);
      s0 = wmma_bf16(qa0, kb00, s0);
      s0 = wmma_bf16(qa1, kb01, s0);
      s1 = wmma_bf16(qa0, kb10, s1);
      s1 = wmma_bf16(qa1, kb11, s1);
    }

    // ---- ALiBi + online softmax ----
    float k0pos = (float)(kt + n);
    float k1pos = (float)(kt + 16 + n);
#pragma unroll
    for (int i = 0; i < 8; ++i) {
      float qpos = (float)(q0 + i + mhi);
      float a0 = s0[i] * inv_sqrt_d + slope * (k0pos - qpos);
      float a1 = s1[i] * inv_sqrt_d + slope * (k1pos - qpos);
      float v = fmaxf(a0, a1);
#pragma unroll
      for (int off = 8; off; off >>= 1) v = fmaxf(v, __shfl_xor(v, off, 32));
      float mnew = fmaxf(mrow[i], v);
      float corr = __expf(mrow[i] - mnew);
      float p0 = __expf(a0 - mnew);
      float p1 = __expf(a1 - mnew);
      float rs = p0 + p1;
#pragma unroll
      for (int off = 8; off; off >>= 1) rs += __shfl_xor(rs, off, 32);
      lrow[i] = lrow[i] * corr + rs;
      mrow[i] = mnew;
#pragma unroll
      for (int j = 0; j < 4; ++j) acc[j][i] *= corr;
      int m = i + mhi;
      pt[m * 32 + n]      = (__bf16)p0;  // C-layout -> LDS row-major [16][32]
      pt[m * 32 + 16 + n] = (__bf16)p1;
    }
    asm volatile("s_wait_dscnt 0" ::: "memory");  // per-wave P visibility

    // ---- P(16x32) x V(32x64) from LDS: batch 4 V fragments, 4 WMMAs ----
    {
      v16bf pa = load_a_frag(pt, 0, 32, 0, lane);
      v16bf vb0 = load_b_frag(vs, 0 * 16, 32, 0, lane);
      v16bf vb1 = load_b_frag(vs, 1 * 16, 32, 0, lane);
      v16bf vb2 = load_b_frag(vs, 2 * 16, 32, 0, lane);
      v16bf vb3 = load_b_frag(vs, 3 * 16, 32, 0, lane);
      acc[0] = wmma_bf16(pa, vb0, acc[0]);
      acc[1] = wmma_bf16(pa, vb1, acc[1]);
      acc[2] = wmma_bf16(pa, vb2, acc[2]);
      acc[3] = wmma_bf16(pa, vb3, acc[3]);
    }
    __syncthreads();  // all waves done reading buf before it is re-staged
  }

  // ---- normalize + write head output (concat: col = head*64 + d) ----
#pragma unroll
  for (int j = 0; j < 4; ++j)
#pragma unroll
    for (int i = 0; i < 8; ++i) {
      float o = acc[j][i] / lrow[i];
      Ob[((size_t)b * SQ + q0 + i + mhi) * HQ + head * DQ + j * 16 + n] =
          (__bf16)o;
    }
#undef STAGE_TILES
}

// ---------------- output projection: out = Ob @ Wo^T + bias ----------------
__global__ __launch_bounds__(256) void out_proj_kernel(
    const __bf16* __restrict__ Ob, const __bf16* __restrict__ wob,
    const float* __restrict__ bias, float* __restrict__ out) {
  int lane = threadIdx.x & 31;
  int wave = threadIdx.x >> 5;
  int grow = blockIdx.x * 128 + wave * 16;
  int n0   = blockIdx.y * 64;

  v8f acc[4] = {};
  for (int kk = 0; kk < HQ; kk += 32) {
    v16bf a  = load_a_frag(Ob, grow, HQ, kk, lane);
    v16bf b0 = load_b_frag(wob, n0 + 0 * 16, HQ, kk, lane);
    v16bf b1 = load_b_frag(wob, n0 + 1 * 16, HQ, kk, lane);
    v16bf b2 = load_b_frag(wob, n0 + 2 * 16, HQ, kk, lane);
    v16bf b3 = load_b_frag(wob, n0 + 3 * 16, HQ, kk, lane);
    acc[0] = wmma_bf16(a, b0, acc[0]);
    acc[1] = wmma_bf16(a, b1, acc[1]);
    acc[2] = wmma_bf16(a, b2, acc[2]);
    acc[3] = wmma_bf16(a, b3, acc[3]);
  }

  int n   = lane & 15;
  int mhi = (lane >> 4) << 3;
#pragma unroll
  for (int j = 0; j < 4; ++j) {
    float bj = bias[n0 + j * 16 + n];
#pragma unroll
    for (int i = 0; i < 8; ++i)
      out[(size_t)(grow + i + mhi) * HQ + (n0 + j * 16 + n)] = acc[j][i] + bj;
  }
}

extern "C" void kernel_launch(void* const* d_in, const int* in_sizes, int n_in,
                              void* d_out, int out_size, void* d_ws,
                              size_t ws_size, hipStream_t stream) {
  (void)in_sizes; (void)n_in; (void)out_size; (void)ws_size;
  const float* x  = (const float*)d_in[0];
  const float* Wq = (const float*)d_in[1];
  const float* Wk = (const float*)d_in[2];
  const float* Wv = (const float*)d_in[3];
  const float* Wo = (const float*)d_in[4];
  const float* bo = (const float*)d_in[5];
  float* out = (float*)d_out;

  const size_t XN = (size_t)BQ * SQ * HQ;    // 8,388,608
  const size_t WN = (size_t)NHQ * DQ * HQ;   // 1,048,576
  const size_t ON = (size_t)HQ * HQ;         // 1,048,576

  __bf16* xb  = (__bf16*)d_ws;
  __bf16* wqb = xb  + XN;
  __bf16* wkb = wqb + WN;
  __bf16* wvb = wkb + WN;
  __bf16* wob = wvb + WN;
  __bf16* Qb  = wob + ON;
  __bf16* Kb  = Qb  + XN;
  __bf16* Vt  = Kb  + XN;
  __bf16* Ob  = Vt  + XN;   // total ~92 MB bf16 scratch

  cvt_f32_bf16_kernel<<<(int)(XN / 4 / 256), 256, 0, stream>>>(x,  xb,  (int)(XN / 4));
  cvt_f32_bf16_kernel<<<(int)(WN / 4 / 256), 256, 0, stream>>>(Wq, wqb, (int)(WN / 4));
  cvt_f32_bf16_kernel<<<(int)(WN / 4 / 256), 256, 0, stream>>>(Wk, wkb, (int)(WN / 4));
  cvt_f32_bf16_kernel<<<(int)(WN / 4 / 256), 256, 0, stream>>>(Wv, wvb, (int)(WN / 4));
  cvt_f32_bf16_kernel<<<(int)(ON / 4 / 256), 256, 0, stream>>>(Wo, wob, (int)(ON / 4));

  // fused QKV projection: 64 row tiles x 16 heads
  qkv_proj_kernel<<<dim3(64, NHQ), 256, 0, stream>>>(xb, wqb, wkb, wvb,
                                                     Qb, Kb, Vt);
  // attention: (S/128, NH, B)
  attn_alibi_kernel<<<dim3(SQ / 128, NHQ, BQ), 256, 0, stream>>>(Qb, Kb, Vt, Ob);
  // output projection
  out_proj_kernel<<<dim3(64, HQ / 64), 256, 0, stream>>>(Ob, wob, bo, out);
}
